// AttentionPolicyHead_44848048505319
// MI455X (gfx1250) — matrix operand order repaired and checked by reading, not compiled, not run on hardware
//
#include <hip/hip_runtime.h>
#include <hip/hip_bf16.h>

// Fused AttentionPolicyHead for gfx1250 (MI455X).
// One workgroup (256 threads = 8 wave32) per batch element; all intermediates
// in LDS (~288KB of 320KB/WGP); GEMMs via v_wmma_f32_16x16x32_f16; the x[b]
// slab (60.5KB, the only HBM-heavy input) is DMA'd to LDS by the Tensor Data
// Mover (tensor_load_to_lds) so the hex gather never does strided HBM reads.

typedef __attribute__((ext_vector_type(16))) _Float16 v16h;
typedef __attribute__((ext_vector_type(8)))  float    v8f;
typedef unsigned int u32x4 __attribute__((ext_vector_type(4)));
typedef unsigned int u32x8 __attribute__((ext_vector_type(8)));

#define NTHREADS 256
#define WAVES 8

__device__ __forceinline__ v8f wmma_f16(v16h a, v16h b, v8f c) {
  // (neg_a, A, neg_b, B, c_mod, C, reuse_a, reuse_b)
  return __builtin_amdgcn_wmma_f32_16x16x32_f16(false, a, false, b, (short)0, c,
                                                false, false);
}

// ---- TDM: 1-D contiguous global->LDS DMA (D# per ISA 08_async_tensor §8) ---
// group0: [1:0]=count=1, [63:32]=lds_addr, [120:64]=global_addr, [127:126]=2
// group1: [17:16]=data_size(2 -> 4B), tensor_dim0[79:48], tensor_dim1[111:80],
//         tile_dim0[127:112], tile_dim1[143:128], tensor_dim0_stride[207:160]
__device__ __forceinline__ void tdm_load_1d_f32(const float* gptr,
                                                void* lds_ptr,
                                                unsigned nelem) {
  unsigned long long ga = (unsigned long long)(uintptr_t)gptr;
  unsigned lds_addr = (unsigned)(uintptr_t)lds_ptr;  // LDS aperture: addr[31:0]
  u32x4 g0;
  g0.x = 1u;                                               // count=1
  g0.y = lds_addr;
  g0.z = (unsigned)ga;
  g0.w = ((unsigned)(ga >> 32) & 0x01FFFFFFu) | (2u << 30); // type=2 (image)
  u32x8 g1;
  g1.s0 = 2u << 16;                          // data_size = 4 bytes
  g1.s1 = (nelem & 0xFFFFu) << 16;           // tensor_dim0 lo16
  g1.s2 = ((nelem >> 16) & 0xFFFFu) | (1u << 16);  // dim0 hi16 | tensor_dim1=1
  g1.s3 = (nelem & 0xFFFFu) << 16;           // tile_dim0 = nelem (<=65535)
  g1.s4 = 1u;                                // tile_dim1 = 1
  g1.s5 = nelem;                             // tensor_dim0_stride lo32
  g1.s6 = 0u;
  g1.s7 = 0u;
  asm volatile("tensor_load_to_lds %0, %1" : : "s"(g0), "s"(g1) : "memory");
}

// A fragment: 16x32 tile at (m0,k0) of row-major src (stride ld), contiguous K order:
// lanes 0-15 -> K 0..15, lanes 16-31 -> K 16..31; VGPR v holds a K pair.
template <typename T>
__device__ __forceinline__ v16h frag_a(const T* __restrict__ src, int ld,
                                       int m0, int k0, int lane) {
  const int m  = m0 + (lane & 15);
  const int kb = (lane & 16) ? 16 : 0;
  const T* row = src + m * ld + k0 + kb;
  v16h a;
#pragma unroll
  for (int v = 0; v < 8; ++v) {
    const int k = 2 * v;  // 0,2,4,...,14
    a[2 * v]     = (_Float16)row[k];
    a[2 * v + 1] = (_Float16)row[k + 1];
  }
  return a;
}

// A fragment with K masked to kmax (zero-padded); k0 == 0.
template <typename T>
__device__ __forceinline__ v16h frag_a_mask(const T* __restrict__ src, int ld,
                                            int m0, int lane, int kmax) {
  const int m  = m0 + (lane & 15);
  const int kb = (lane & 16) ? 16 : 0;
  v16h a;
#pragma unroll
  for (int v = 0; v < 8; ++v) {
    const int k = kb + 2 * v;
    a[2 * v]     = (k     < kmax) ? (_Float16)src[m * ld + k]     : (_Float16)0.f;
    a[2 * v + 1] = (k + 1 < kmax) ? (_Float16)src[m * ld + k + 1] : (_Float16)0.f;
  }
  return a;
}

// B fragment from generic row-major [K][N] f32 (activations): lanes 0-15 hold
// K 0..15 (VGPR v -> K 2v,2v+1), lanes 16-31 hold K 16..31.
template <typename T>
__device__ __forceinline__ v16h frag_b(const T* __restrict__ src, int ld,
                                       int k0, int n0, int lane) {
  const int n  = n0 + (lane & 15);
  const int kb = (lane & 16) ? 16 : 0;
  v16h b;
#pragma unroll
  for (int v = 0; v < 8; ++v) {
    const int k = k0 + kb + 2 * v;
    b[2 * v]     = (_Float16)src[k * ld + n];
    b[2 * v + 1] = (_Float16)src[(k + 1) * ld + n];
  }
  return b;
}

// B fragment from transposed storage: B[k][n] = base[n*rs + k], K masked.
__device__ __forceinline__ v16h frag_bT_mask(const float* __restrict__ base,
                                             int rs, int n0, int lane, int kmax) {
  const int n  = n0 + (lane & 15);
  const int kb = (lane & 16) ? 16 : 0;
  v16h b;
#pragma unroll
  for (int v = 0; v < 8; ++v) {
    const int k = kb + 2 * v;
    b[2 * v]     = (k     < kmax) ? (_Float16)base[n * rs + k]     : (_Float16)0.f;
    b[2 * v + 1] = (k + 1 < kmax) ? (_Float16)base[n * rs + k + 1] : (_Float16)0.f;
  }
  return b;
}

// B fragment from pre-swizzled weight staging: each lane's 16 halves are
// contiguous (32B) -> 2x ds_load_b128.
__device__ __forceinline__ v16h frag_b_sw(const _Float16* __restrict__ Wsw,
                                          int NT, int kt, int nt, int lane) {
  const _Float16* p = Wsw + (((kt * NT + nt) * 32 + lane) << 4);
  v16h b;
#pragma unroll
  for (int e = 0; e < 16; ++e) b[e] = p[e];
  return b;
}

__global__ __launch_bounds__(NTHREADS) void policy_head_fused(
    const float* __restrict__ x,
    const float* __restrict__ embed_w, const float* __restrict__ embed_b,
    const float* __restrict__ pos,
    const float* __restrict__ in_w,  const float* __restrict__ in_b,
    const float* __restrict__ out_w, const float* __restrict__ out_b,
    const float* __restrict__ ln1_g, const float* __restrict__ ln1_b,
    const float* __restrict__ ln2_g, const float* __restrict__ ln2_b,
    const float* __restrict__ ff1_w, const float* __restrict__ ff1_b,
    const float* __restrict__ ff2_w, const float* __restrict__ ff2_b,
    const float* __restrict__ from_w, const float* __restrict__ from_b,
    const float* __restrict__ to_w,   const float* __restrict__ to_b,
    const float* __restrict__ move_bias,
    const int* __restrict__ move_from, const int* __restrict__ move_to,
    float* __restrict__ out) {
  // LDS: 96*192 + 96*64 + 96*64 + 4*96*96 f32 + 12288 f16 = 294912 B
  extern __shared__ char smem[];
  float* X  = (float*)smem;              // 96x192 f32: qkv / ff1 act / pos
  float* H  = X + 96 * 192;              // 96x64 f32: residual stream h
  float* Nb = H + 96 * 64;               // 96x64 f32: hn / attention-out o
  float* S  = Nb + 96 * 64;              // 4x96x96 f32: scores; later from/to
  _Float16* W = (_Float16*)(S + 4 * 96 * 96);   // 12288 f16 swizzled weights
  _Float16* F = (_Float16*)S;            // alias: 96x128 f16 feats (S[0:24576))
  float* slab = (float*)((char*)S + 24576);     // alias: x[b] slab (61952 B)

  __shared__ int hexr[91];
  __shared__ int hexc[91];

  const int b    = blockIdx.x;
  const int tid  = threadIdx.x;
  const int lane = tid & 31;
  const int wv   = tid >> 5;

  // Stage weight (N=O rows, K cols, row-major) into swizzled B-fragment form.
  auto stage_sw = [&](const float* __restrict__ w, int Ntot, int Ktot,
                      _Float16* dst) {
    const int NT = Ntot >> 4;
    const int total = Ntot * Ktot;
    for (int d = tid; d < total; d += NTHREADS) {
      const int e  = d & 15;
      const int ln = (d >> 4) & 31;
      const int tile = d >> 9;
      const int kt = tile / NT, nt = tile - kt * NT;
      const int n = nt * 16 + (ln & 15);
      const int k = kt * 32 + ((ln & 16) ? 16 : 0) + e;
      dst[d] = (_Float16)w[n * Ktot + k];
    }
  };
  auto layernorm = [&](const float* __restrict__ g, const float* __restrict__ bb) {
    if (tid < 96) {
      const float* hr = H + tid * 64;
      float mean = 0.f;
      for (int d = 0; d < 64; ++d) mean += hr[d];
      mean *= (1.f / 64.f);
      float var = 0.f;
      for (int d = 0; d < 64; ++d) { float t = hr[d] - mean; var += t * t; }
      var *= (1.f / 64.f);
      const float rs = rsqrtf(var + 1e-5f);
      for (int d = 0; d < 64; ++d)
        Nb[tid * 64 + d] = (hr[d] - mean) * rs * g[d] + bb[d];
    }
  };

  // ---- kick off TDM DMA of the whole x[b] slab (128*121 f32, contiguous) ---
  if (wv == 0) {
    tdm_load_1d_f32(x + (size_t)b * 128 * 121, slab, 128 * 121);
  }

  // ---- overlap: hex table, pos staging (padded), embed weights -------------
  if (tid < 91) {
    int cnt = 0, rr = 0, cc = 0;
    for (int q = -5; q <= 5; ++q)
      for (int r = -5; r <= 5; ++r) {
        int s = q + r;
        int a0 = q < 0 ? -q : q, a1 = r < 0 ? -r : r, a2 = s < 0 ? -s : s;
        int mx = a0 > a1 ? a0 : a1; mx = mx > a2 ? mx : a2;
        if (mx <= 5) { if (cnt == tid) { rr = r + 5; cc = q + 5; } ++cnt; }
      }
    hexr[tid] = rr; hexc[tid] = cc;
  }
  for (int i = tid; i < 96 * 64; i += NTHREADS)     // zero-padded pos in X
    X[i] = (i < 91 * 64) ? pos[i] : 0.f;
  stage_sw(embed_w, 64, 128, W);

  __builtin_amdgcn_s_wait_tensorcnt(0);             // slab resident
  __syncthreads();

  // ---- gather feats from LDS slab -> F[96][128] f16 (rows 91..95 zero) -----
  for (int i = tid; i < 96 * 128; i += NTHREADS) {
    int c = i >> 7, f = i & 127;
    float v = (c < 91) ? slab[f * 121 + hexr[c] * 11 + hexc[c]] : 0.f;
    F[i] = (_Float16)v;
  }
  __syncthreads();

  // ---- embed: H = F @ embed_w^T + embed_b + pos   (M96 N64 K128) ----------
  for (int job = wv; job < 24; job += WAVES) {
    const int m0 = (job >> 2) * 16, nt = job & 3, n0 = nt * 16;
    v8f acc = {};
#pragma unroll
    for (int kk = 0; kk < 4; ++kk)
      acc = wmma_f16(frag_a(F, 128, m0, kk * 32, lane),
                     frag_b_sw(W, 4, kk, nt, lane), acc);
    const int n = n0 + (lane & 15), mb = (lane & 16) ? 8 : 0;
    const float bn = embed_b[n];
#pragma unroll
    for (int r = 0; r < 8; ++r) {
      const int m = m0 + mb + r;
      H[m * 64 + n] = acc[r] + bn + X[m * 64 + n];
    }
  }
  __syncthreads();

  // ---- LN1 -> Nb ; stage in_w (K64 x N192) --------------------------------
  layernorm(ln1_g, ln1_b);
  stage_sw(in_w, 192, 64, W);
  __syncthreads();

  // ---- QKV: X = Nb @ in_w^T + in_b   (M96 N192 K64) -----------------------
  for (int job = wv; job < 72; job += WAVES) {
    const int m0 = (job / 12) * 16, nt = job % 12, n0 = nt * 16;
    v8f acc = {};
    acc = wmma_f16(frag_a(Nb, 64, m0, 0, lane),  frag_b_sw(W, 12, 0, nt, lane), acc);
    acc = wmma_f16(frag_a(Nb, 64, m0, 32, lane), frag_b_sw(W, 12, 1, nt, lane), acc);
    const int n = n0 + (lane & 15), mb = (lane & 16) ? 8 : 0;
    const float bn = in_b[n];
#pragma unroll
    for (int r = 0; r < 8; ++r) X[(m0 + mb + r) * 192 + n] = acc[r] + bn;
  }
  __syncthreads();

  // ---- scores[h] = (q @ k^T) / 4   (per head M96 N96 K16->pad32) ----------
  for (int job = wv; job < 144; job += WAVES) {
    const int hh = job / 36, t = job % 36;
    const int m0 = (t / 6) * 16, n0 = (t % 6) * 16;
    v8f acc = {};
    acc = wmma_f16(frag_a_mask(X + hh * 16, 192, m0, lane, 16),
                   frag_bT_mask(X + 64 + hh * 16, 192, n0, lane, 16), acc);
    const int n = n0 + (lane & 15), mb = (lane & 16) ? 8 : 0;
    float* Sh = S + hh * 96 * 96;
#pragma unroll
    for (int r = 0; r < 8; ++r) Sh[(m0 + mb + r) * 96 + n] = acc[r] * 0.25f;
  }
  stage_sw(out_w, 64, 64, W);  // out-proj weights (W free since QKV)
  __syncthreads();

  // ---- softmax over keys (j<91), in place ---------------------------------
  for (int rrow = tid; rrow < 4 * 96; rrow += NTHREADS) {
    float* row = S + (rrow / 96) * 96 * 96 + (rrow % 96) * 96;
    float mx = -1e30f;
    for (int j = 0; j < 91; ++j) mx = fmaxf(mx, row[j]);
    float sum = 0.f;
    for (int j = 0; j < 91; ++j) { float e = __expf(row[j] - mx); row[j] = e; sum += e; }
    const float inv = 1.f / sum;
    for (int j = 0; j < 91; ++j) row[j] *= inv;
    for (int j = 91; j < 96; ++j) row[j] = 0.f;
  }
  __syncthreads();

  // ---- o = attn @ v -> Nb   (per head M96 N16 K96) ------------------------
  for (int job = wv; job < 24; job += WAVES) {
    const int hh = job / 6, m0 = (job % 6) * 16;
    const float* Sh = S + hh * 96 * 96;
    const float* Vh = X + 128 + hh * 16;
    v8f acc = {};
#pragma unroll
    for (int kk = 0; kk < 3; ++kk)
      acc = wmma_f16(frag_a(Sh, 96, m0, kk * 32, lane),
                     frag_b(Vh, 192, kk * 32, 0, lane), acc);
    const int n = lane & 15, mb = (lane & 16) ? 8 : 0;
#pragma unroll
    for (int r = 0; r < 8; ++r) Nb[(m0 + mb + r) * 64 + hh * 16 + n] = acc[r];
  }
  __syncthreads();

  // ---- H += o @ out_w^T + out_b   (M96 N64 K64) ---------------------------
  for (int job = wv; job < 24; job += WAVES) {
    const int m0 = (job >> 2) * 16, nt = job & 3, n0 = nt * 16;
    v8f acc = {};
    acc = wmma_f16(frag_a(Nb, 64, m0, 0, lane),  frag_b_sw(W, 4, 0, nt, lane), acc);
    acc = wmma_f16(frag_a(Nb, 64, m0, 32, lane), frag_b_sw(W, 4, 1, nt, lane), acc);
    const int n = n0 + (lane & 15), mb = (lane & 16) ? 8 : 0;
    const float bn = out_b[n];
#pragma unroll
    for (int r = 0; r < 8; ++r) H[(m0 + mb + r) * 64 + n] += acc[r] + bn;
  }
  __syncthreads();

  // ---- LN2 -> Nb ; stage ff1_w (K64 x N128) -------------------------------
  layernorm(ln2_g, ln2_b);
  stage_sw(ff1_w, 128, 64, W);
  __syncthreads();

  // ---- FF1: X = relu(Nb @ ff1_w^T + ff1_b)   (M96 N128 K64) ---------------
  for (int job = wv; job < 48; job += WAVES) {
    const int m0 = (job >> 3) * 16, nt = job & 7, n0 = nt * 16;
    v8f acc = {};
    acc = wmma_f16(frag_a(Nb, 64, m0, 0, lane),  frag_b_sw(W, 8, 0, nt, lane), acc);
    acc = wmma_f16(frag_a(Nb, 64, m0, 32, lane), frag_b_sw(W, 8, 1, nt, lane), acc);
    const int n = n0 + (lane & 15), mb = (lane & 16) ? 8 : 0;
    const float bn = ff1_b[n];
#pragma unroll
    for (int r = 0; r < 8; ++r)
      X[(m0 + mb + r) * 128 + n] = fmaxf(acc[r] + bn, 0.f);
  }
  __syncthreads();
  stage_sw(ff2_w, 64, 128, W);  // K128 x N64
  __syncthreads();

  // ---- FF2: H += X @ ff2_w^T + ff2_b   (M96 N64 K128) ---------------------
  for (int job = wv; job < 24; job += WAVES) {
    const int m0 = (job >> 2) * 16, nt = job & 3, n0 = nt * 16;
    v8f acc = {};
#pragma unroll
    for (int kk = 0; kk < 4; ++kk)
      acc = wmma_f16(frag_a(X, 128, m0, kk * 32, lane),
                     frag_b_sw(W, 4, kk, nt, lane), acc);
    const int n = n0 + (lane & 15), mb = (lane & 16) ? 8 : 0;
    const float bn = ff2_b[n];
#pragma unroll
    for (int r = 0; r < 8; ++r) H[(m0 + mb + r) * 64 + n] += acc[r] + bn;
  }
  __syncthreads();

  // ---- from/to projections into S (stride 65: bank-conflict-free dots) ----
  stage_sw(from_w, 64, 64, W);
  stage_sw(to_w, 64, 64, W + 4096);
  __syncthreads();
  for (int job = wv; job < 48; job += WAVES) {
    const int which = job / 24, t = job % 24;
    const int m0 = (t >> 2) * 16, nt = t & 3, n0 = nt * 16;
    const _Float16* Ww = W + which * 4096;
    const float* bb = which ? to_b : from_b;
    float* dst = S + which * 96 * 65;
    v8f acc = {};
    acc = wmma_f16(frag_a(H, 64, m0, 0, lane),  frag_b_sw(Ww, 4, 0, nt, lane), acc);
    acc = wmma_f16(frag_a(H, 64, m0, 32, lane), frag_b_sw(Ww, 4, 1, nt, lane), acc);
    const int n = n0 + (lane & 15), mb = (lane & 16) ? 8 : 0;
    const float bn = bb[n];
#pragma unroll
    for (int r = 0; r < 8; ++r) dst[(m0 + mb + r) * 65 + n] = acc[r] + bn;
  }
  __syncthreads();

  // ---- move logits: dot(from[move_from], to[move_to]) + bias --------------
  const float* FF = S;
  const float* TT = S + 96 * 65;
  float* outp = out + (size_t)b * 4206;
  for (int m = tid; m < 4206; m += NTHREADS) {
    const float* fv = FF + move_from[m] * 65;
    const float* tv = TT + move_to[m] * 65;
    float acc = move_bias[m];
#pragma unroll 8
    for (int d = 0; d < 64; ++d) acc += fv[d] * tv[d];
    outp[m] = acc;
  }
}

extern "C" void kernel_launch(void* const* d_in, const int* in_sizes, int n_in,
                              void* d_out, int out_size, void* d_ws, size_t ws_size,
                              hipStream_t stream) {
  const float* x        = (const float*)d_in[0];
  const float* embed_w  = (const float*)d_in[1];
  const float* embed_b  = (const float*)d_in[2];
  const float* pos      = (const float*)d_in[3];
  const float* in_w     = (const float*)d_in[4];
  const float* in_b     = (const float*)d_in[5];
  const float* out_w    = (const float*)d_in[6];
  const float* out_b    = (const float*)d_in[7];
  const float* ln1_g    = (const float*)d_in[8];
  const float* ln1_b    = (const float*)d_in[9];
  const float* ln2_g    = (const float*)d_in[10];
  const float* ln2_b    = (const float*)d_in[11];
  const float* ff1_w    = (const float*)d_in[12];
  const float* ff1_b    = (const float*)d_in[13];
  const float* ff2_w    = (const float*)d_in[14];
  const float* ff2_b    = (const float*)d_in[15];
  const float* from_w   = (const float*)d_in[16];
  const float* from_b   = (const float*)d_in[17];
  const float* to_w     = (const float*)d_in[18];
  const float* to_b     = (const float*)d_in[19];
  const float* mv_bias  = (const float*)d_in[20];
  const int*   mv_from  = (const int*)d_in[21];
  const int*   mv_to    = (const int*)d_in[22];
  float* out = (float*)d_out;

  const int B = in_sizes[0] / (128 * 121);
  const size_t smem = (size_t)(96 * 192 + 96 * 64 + 96 * 64 + 4 * 96 * 96) * 4
                      + 12288 * 2;  // 294912 bytes
  (void)hipFuncSetAttribute((const void*)policy_head_fused,
                            hipFuncAttributeMaxDynamicSharedMemorySize,
                            (int)smem);
  policy_head_fused<<<B, NTHREADS, smem, stream>>>(
      x, embed_w, embed_b, pos, in_w, in_b, out_w, out_b,
      ln1_g, ln1_b, ln2_g, ln2_b, ff1_w, ff1_b, ff2_w, ff2_b,
      from_w, from_b, to_w, to_b, mv_bias, mv_from, mv_to, out);
}
